// UnfoldAndLinear_84739704750320
// MI455X (gfx1250) — 8-bit matrix operand order repaired and checked by reading, compiled, not run on hardware
//
#include <hip/hip_runtime.h>
#include <stdint.h>

typedef int v8i __attribute__((ext_vector_type(8)));

#define C_IN   320
#define K_REAL 2880
#define K_PAD  2944
#define O_PAD  384
#define RANK   32
#define HW     64
#define NTOK   16384
#define ROWB   2960

__global__ __launch_bounds__(256) void quant_w_kernel(const float* __restrict__ w,
                                                      int8_t* __restrict__ qw,
                                                      float* __restrict__ sw) {
    __shared__ float red[256];
    const int o = blockIdx.x;
    const int tid = threadIdx.x;
    const float* wr = w + (size_t)o * K_PAD;

    float amax = 0.0f;
    for (int k = tid; k < K_PAD; k += 256) amax = fmaxf(amax, fabsf(wr[k]));
    red[tid] = amax;
    __syncthreads();
    for (int s = 128; s > 0; s >>= 1) {
        if (tid < s) red[tid] = fmaxf(red[tid], red[tid + s]);
        __syncthreads();
    }
    const float sc = fmaxf(red[0] * (1.0f / 7.0f), 1e-8f);
    if (tid == 0) sw[o] = sc;
    for (int k = tid; k < K_PAD; k += 256) {
        float q = rintf(wr[k] / sc);
        q = fminf(fmaxf(q, -8.0f), 7.0f);
        qw[(size_t)o * K_PAD + k] = (int8_t)q;
    }
}

__global__ __launch_bounds__(256) void fused_kernel(const float* __restrict__ x,
                                                    const int8_t* __restrict__ qw,
                                                    const float* __restrict__ sw,
                                                    const float* __restrict__ proj_down,
                                                    const float* __restrict__ proj_up,
                                                    const float* __restrict__ bias,
                                                    float* __restrict__ out) {
    __shared__ __align__(16) int8_t qA[16 * ROWB];
    __shared__ float lora1_s[16 * RANK];
    __shared__ float sx_s[16];

    const int tid = threadIdx.x;
    const int tok = tid & 15;
    const int g   = tid >> 4;
    const int tok_base = blockIdx.x * 16;

    const int tglob = tok_base + tok;
    const int n_img = tglob >> 12;
    const int l     = tglob & 4095;
    const int oh    = l >> 6;
    const int ow    = l & 63;
    const float* xn = x + (size_t)n_img * C_IN * HW * HW;

    float* scr = (float*)qA;
    float acc[RANK];
#pragma unroll
    for (int r = 0; r < RANK; ++r) acc[r] = 0.0f;
    float amax = 0.0f;

    for (int k = g; k < K_REAL; k += 16) {
        const int c   = k / 9;
        const int rem = k - c * 9;
        const int di  = rem / 3;
        const int dj  = rem - di * 3;
        const int ih  = oh - 1 + di;
        const int iw  = ow - 1 + dj;
        float v = 0.0f;
        if ((unsigned)ih < HW && (unsigned)iw < HW)
            v = xn[((size_t)c * HW + ih) * HW + iw];
        amax = fmaxf(amax, fabsf(v));
        const float4* pd4 = (const float4*)(proj_down + (size_t)k * RANK);
#pragma unroll
        for (int rr = 0; rr < RANK / 4; ++rr) {
            const float4 p = pd4[rr];
            acc[4 * rr + 0] = fmaf(v, p.x, acc[4 * rr + 0]);
            acc[4 * rr + 1] = fmaf(v, p.y, acc[4 * rr + 1]);
            acc[4 * rr + 2] = fmaf(v, p.z, acc[4 * rr + 2]);
            acc[4 * rr + 3] = fmaf(v, p.w, acc[4 * rr + 3]);
        }
    }
    {
        float* my = scr + ((g * 16 + tok) * RANK);
#pragma unroll
        for (int r = 0; r < RANK; ++r) my[r] = acc[r];
        scr[8192 + g * 16 + tok] = amax;
    }
    __syncthreads();

    {
        const int p0 = tid * 2;
#pragma unroll
        for (int pp = 0; pp < 2; ++pp) {
            const int p = p0 + pp;
            const int t = p >> 5;
            const int r = p & 31;
            float s = 0.0f;
#pragma unroll
            for (int gg = 0; gg < 16; ++gg) s += scr[(gg * 16 + t) * RANK + r];
            lora1_s[t * RANK + r] = s;
        }
        if (tid < 16) {
            float m = 0.0f;
#pragma unroll
            for (int gg = 0; gg < 16; ++gg) m = fmaxf(m, scr[8192 + gg * 16 + tid]);
            sx_s[tid] = fmaxf(m * (1.0f / 7.0f), 1e-8f);
        }
    }
    __syncthreads();

    {
        const float sxv = sx_s[tok];
        int k = g;
        for (int it = 0; it < 184; ++it, k += 16) {
            float v = 0.0f;
            if (k < K_REAL) {
                const int c   = k / 9;
                const int rem = k - c * 9;
                const int di  = rem / 3;
                const int dj  = rem - di * 3;
                const int ih  = oh - 1 + di;
                const int iw  = ow - 1 + dj;
                if ((unsigned)ih < HW && (unsigned)iw < HW)
                    v = xn[((size_t)c * HW + ih) * HW + iw];
            }
            float q = fminf(fmaxf(rintf(v / sxv), -8.0f), 7.0f);
            qA[tok * ROWB + k] = (int8_t)q;
        }
    }
    __syncthreads();

    const int lane = tid & 31;
    const int wv   = tid >> 5;
    const int hi   = lane >> 4;
    const int lm   = lane & 15;

    v8i cacc[3];
#pragma unroll
    for (int t = 0; t < 3; ++t)
#pragma unroll
        for (int j = 0; j < 8; ++j) cacc[t][j] = 0;

    const int8_t* a_row = &qA[lm * ROWB + hi * 8];

    for (int k0 = 0; k0 < K_PAD; k0 += 64) {
        const int* ap = (const int*)(a_row + k0);
        v8i a;
        a[0] = ap[0];  a[1] = ap[1];
        a[2] = ap[4];  a[3] = ap[5];
        a[4] = ap[8];  a[5] = ap[9];
        a[6] = ap[12]; a[7] = ap[13];
#pragma unroll
        for (int t = 0; t < 3; ++t) {
            const int n_base = (wv * 3 + t) * 16;
            const int2* bp = (const int2*)(qw + (size_t)(n_base + lm) * K_PAD + k0 + hi * 8);
            const int2 b0 = bp[0];
            const int2 b1 = bp[2];
            const int2 b2 = bp[4];
            const int2 b3 = bp[6];
            v8i b;
            b[0] = b0.x; b[1] = b0.y; b[2] = b1.x; b[3] = b1.y;
            b[4] = b2.x; b[5] = b2.y; b[6] = b3.x; b[7] = b3.y;
            cacc[t] = __builtin_amdgcn_wmma_i32_16x16x64_iu8(
                 true, a,  true, b, cacc[t],
                 false,  false);
        }
    }

#pragma unroll
    for (int t = 0; t < 3; ++t) {
        const int n_base = (wv * 3 + t) * 16;
        const int o = n_base + lm;
        const float swv = sw[o];
        const float bv  = bias[o];
        float f[8];
#pragma unroll
        for (int j = 0; j < 8; ++j) {
            const int M = j + hi * 8;
            f[j] = (float)cacc[t][j] * sx_s[M] * swv + bv;
        }
        for (int r = 0; r < RANK; ++r) {
            const float puv = proj_up[(size_t)r * O_PAD + o];
#pragma unroll
            for (int j = 0; j < 8; ++j) {
                const int M = j + hi * 8;
                f[j] = fmaf(lora1_s[M * RANK + r], puv, f[j]);
            }
        }
#pragma unroll
        for (int j = 0; j < 8; ++j) {
            const int M  = j + hi * 8;
            const int tg = tok_base + M;
            out[(size_t)tg * O_PAD + o] = f[j];
        }
    }
}

extern "C" void kernel_launch(void* const* d_in, const int* in_sizes, int n_in,
                              void* d_out, int out_size, void* d_ws, size_t ws_size,
                              hipStream_t stream) {
    const float* x  = (const float*)d_in[0];
    const float* w  = (const float*)d_in[1];
    const float* pd = (const float*)d_in[2];
    const float* pu = (const float*)d_in[3];
    const float* b  = (const float*)d_in[4];
    float* out = (float*)d_out;

    int8_t* qw = (int8_t*)d_ws;
    float*  sw = (float*)((char*)d_ws + (size_t)O_PAD * K_PAD);

    quant_w_kernel<<<O_PAD, 256, 0, stream>>>(w, qw, sw);
    fused_kernel<<<NTOK / 16, 256, 0, stream>>>(x, qw, sw, pd, pu, b, out);
}
